// BidirectionalMambaBlock_29583734735072
// MI455X (gfx1250) — matrix operand order repaired and checked by reading, not compiled, not run on hardware
//
#include <hip/hip_runtime.h>
#include <hip/hip_bf16.h>

#define D_MODEL 512
#define D_STATE 16
#define D_INNER 1024
#define DT_RANK 32
#define NBATCH  2
#define SEQ     2048
#define MROWS   (NBATCH * SEQ)   // 4096

typedef __attribute__((ext_vector_type(16))) __bf16 v16bf;
typedef __attribute__((ext_vector_type(8)))  __bf16 v8bf;
typedef __attribute__((ext_vector_type(8)))  float  v8f;

// ---------------------------------------------------------------- conversions
__global__ void cvt_f32_bf16(const float* __restrict__ in, __bf16* __restrict__ out, int n) {
    int i = blockIdx.x * 256 + threadIdx.x;
    if (i < n) out[i] = (__bf16)in[i];
}

// x -> bf16 straight copy + bf16 flipped-along-L copy (for the backward branch)
__global__ void prep_x(const float* __restrict__ x, __bf16* __restrict__ xf,
                       __bf16* __restrict__ xb) {
    int i = blockIdx.x * 256 + threadIdx.x;            // over MROWS*D_MODEL
    int d = i & (D_MODEL - 1);
    int m = i >> 9;                                    // D_MODEL = 512 = 2^9
    int b = m >> 11;                                   // SEQ = 2048 = 2^11
    int l = m & (SEQ - 1);
    float v = x[i];
    xf[i] = (__bf16)v;
    xb[((b << 11) + (SEQ - 1 - l)) * D_MODEL + d] = (__bf16)v;
}

// slice proj[:, :32] -> bf16 for the dt GEMM
__global__ void cvt_projdt(const float* __restrict__ proj, __bf16* __restrict__ out) {
    int i = blockIdx.x * 256 + threadIdx.x;            // over MROWS*DT_RANK
    int r = i >> 5, c = i & 31;
    out[i] = (__bf16)proj[r * 64 + c];
}

// ---------------------------------------------------------------- WMMA GEMM
// C[M,N] = A[M,K] * B[N,K]^T  (both row-major, K contiguous; bf16 in, f32 out)
// block = 256 threads = 8 waves; wave -> 16x64 output tile (A-frag reused 4x).
// Software-pipelined: next K-step's fragments are loaded before the current
// step's WMMAs so global loads overlap matrix work (partial loadcnt waits).
__global__ void gemm_bf16_nt(const __bf16* __restrict__ A, int lda,
                             const __bf16* __restrict__ Bw, int ldb,
                             float* __restrict__ C, int ldc, int K) {
    int lane = threadIdx.x & 31;
    int wave = threadIdx.x >> 5;
    int m0 = blockIdx.y * 128 + wave * 16;
    int n0 = blockIdx.x * 64;

    int rA    = m0 + (lane & 15);
    int koffA = (lane >> 4) * 16;     // A 16x32 bf16 fragment K-offset
    int cB    = (lane & 15);
    int koffB = (lane >> 4) * 16;     // B 32x16 bf16 fragment K-offset

    const __bf16* pa = A + (size_t)rA * lda + koffA;
    const __bf16* pb0 = Bw + (size_t)(n0 + 0 * 16 + cB) * ldb + koffB;
    const __bf16* pb1 = Bw + (size_t)(n0 + 1 * 16 + cB) * ldb + koffB;
    const __bf16* pb2 = Bw + (size_t)(n0 + 2 * 16 + cB) * ldb + koffB;
    const __bf16* pb3 = Bw + (size_t)(n0 + 3 * 16 + cB) * ldb + koffB;

    auto ldA = [&](int k0) -> v16bf {
        union { v16bf v; v8bf h[2]; } u;
        u.h[0] = *(const v8bf*)(pa + k0);
        u.h[1] = *(const v8bf*)(pa + k0 + 8);
        return u.v;
    };

    v8f acc[4];
    for (int t = 0; t < 4; ++t)
        for (int r = 0; r < 8; ++r) acc[t][r] = 0.f;

    // prologue: fragments for k0 = 0
    v16bf aC = ldA(0);
    v16bf b0 = *(const v16bf*)(pb0);
    v16bf b1 = *(const v16bf*)(pb1);
    v16bf b2 = *(const v16bf*)(pb2);
    v16bf b3 = *(const v16bf*)(pb3);

    int k0 = 0;
    for (; k0 + 32 < K; k0 += 32) {
        int kn = k0 + 32;
        // issue next-step loads first so they overlap this step's WMMAs
        v16bf aN  = ldA(kn);
        v16bf b0n = *(const v16bf*)(pb0 + kn);
        v16bf b1n = *(const v16bf*)(pb1 + kn);
        v16bf b2n = *(const v16bf*)(pb2 + kn);
        v16bf b3n = *(const v16bf*)(pb3 + kn);
        if (kn + 32 < K) __builtin_prefetch(pa + kn + 32, 0, 1);   // global_prefetch_b8

        acc[0] = __builtin_amdgcn_wmma_f32_16x16x32_bf16(false, aC, false, b0, (short)0, acc[0], false, false);
        acc[1] = __builtin_amdgcn_wmma_f32_16x16x32_bf16(false, aC, false, b1, (short)0, acc[1], false, false);
        acc[2] = __builtin_amdgcn_wmma_f32_16x16x32_bf16(false, aC, false, b2, (short)0, acc[2], false, false);
        acc[3] = __builtin_amdgcn_wmma_f32_16x16x32_bf16(false, aC, false, b3, (short)0, acc[3], false, false);

        aC = aN; b0 = b0n; b1 = b1n; b2 = b2n; b3 = b3n;
    }
    // epilogue: last K-step
    acc[0] = __builtin_amdgcn_wmma_f32_16x16x32_bf16(false, aC, false, b0, (short)0, acc[0], false, false);
    acc[1] = __builtin_amdgcn_wmma_f32_16x16x32_bf16(false, aC, false, b1, (short)0, acc[1], false, false);
    acc[2] = __builtin_amdgcn_wmma_f32_16x16x32_bf16(false, aC, false, b2, (short)0, acc[2], false, false);
    acc[3] = __builtin_amdgcn_wmma_f32_16x16x32_bf16(false, aC, false, b3, (short)0, acc[3], false, false);

    int rowBase = m0 + ((lane >> 4) * 8);
    int col0    = n0 + (lane & 15);
#pragma unroll
    for (int t = 0; t < 4; ++t)
#pragma unroll
        for (int r = 0; r < 8; ++r)
            C[(size_t)(rowBase + r) * ldc + col0 + t * 16] = acc[t][r];
}

// ---------------------------------------------------------------- conv + silu
__global__ void conv_silu(const float* __restrict__ xz, const float* __restrict__ cw,
                          const float* __restrict__ cb, float* __restrict__ uf,
                          __bf16* __restrict__ ub) {
    int idx = blockIdx.x * 256 + threadIdx.x;          // over MROWS*D_INNER
    int d = idx & (D_INNER - 1);
    int m = idx >> 10;
    int b = m >> 11;
    int l = m & (SEQ - 1);
    float acc = cb[d];
#pragma unroll
    for (int k = 0; k < 4; ++k) {
        int ls = l + k - 3;
        if (ls >= 0) acc += xz[(size_t)((b << 11) + ls) * 2048 + d] * cw[d * 4 + k];
    }
    float s = acc / (1.f + __expf(-acc));
    uf[idx] = s;
    ub[idx] = (__bf16)s;
}

// ---------------------------------------------------------------- selective scan
// one thread per channel; h[16] in registers; B/C broadcast via double-buffered
// LDS (one barrier/step); fuses softplus(dt), D-skip, silu(z) gate; bf16 output.
__global__ void scan_kernel(const float* __restrict__ xz,   // dtraw in cols 0..1023, z in 1024..2047
                            const float* __restrict__ proj, // B = cols 32..47, C = 48..63
                            const float* __restrict__ uf,
                            const float* __restrict__ b_dt,
                            const float* __restrict__ A_log,
                            const float* __restrict__ Dskip,
                            __bf16* __restrict__ ybf) {
    int d = blockIdx.x * 256 + threadIdx.x;            // 0..1023
    int b = blockIdx.y;
    float Ad[D_STATE], h[D_STATE];
#pragma unroll
    for (int n = 0; n < D_STATE; ++n) {
        Ad[n] = -__expf(A_log[d * D_STATE + n]);
        h[n] = 0.f;
    }
    float bdt = b_dt[d], dsk = Dskip[d];
    __shared__ float sbc[2][32];
    if (threadIdx.x < 32)
        sbc[0][threadIdx.x] = proj[(size_t)(b << 11) * 64 + 32 + threadIdx.x];
    __syncthreads();
    for (int l = 0; l < SEQ; ++l) {
        int row = (b << 11) + l;
        int cur = l & 1;
        if (l + 1 < SEQ && threadIdx.x < 32)           // stage next step's B/C
            sbc[cur ^ 1][threadIdx.x] = proj[(size_t)(row + 1) * 64 + 32 + threadIdx.x];
        float dtv = xz[(size_t)row * 2048 + d] + bdt;
        dtv = (dtv > 20.f) ? dtv : log1pf(__expf(dtv));   // softplus
        float u  = uf[(size_t)row * D_INNER + d];
        float zz = xz[(size_t)row * 2048 + D_INNER + d];
        float y = 0.f;
#pragma unroll
        for (int n = 0; n < D_STATE; ++n) {
            float dA = __expf(dtv * Ad[n]);
            h[n] = dA * h[n] + (dtv * sbc[cur][n]) * u;
            y += h[n] * sbc[cur][16 + n];
        }
        y += dsk * u;
        float sz = zz / (1.f + __expf(-zz));
        ybf[(size_t)row * D_INNER + d] = (__bf16)(y * sz);
        __syncthreads();
    }
}

// ---------------------------------------------------------------- combine + LN + silu + residual
__global__ void combine_ln(const float* __restrict__ f, const float* __restrict__ bwd,
                           const float* __restrict__ x, const float* __restrict__ g,
                           const float* __restrict__ bb, float* __restrict__ out) {
    int row = blockIdx.x;
    int b = row >> 11, l = row & (SEQ - 1);
    int rb = (b << 11) + (SEQ - 1 - l);
    int tid = threadIdx.x;
    float v0 = 0.5f * (f[(size_t)row * 512 + tid]       + bwd[(size_t)rb * 512 + tid]);
    float v1 = 0.5f * (f[(size_t)row * 512 + tid + 256] + bwd[(size_t)rb * 512 + tid + 256]);
    __shared__ float ssum[256], ssq[256];
    ssum[tid] = v0 + v1;
    ssq[tid]  = v0 * v0 + v1 * v1;
    __syncthreads();
    for (int off = 128; off > 0; off >>= 1) {
        if (tid < off) { ssum[tid] += ssum[tid + off]; ssq[tid] += ssq[tid + off]; }
        __syncthreads();
    }
    float mu  = ssum[0] * (1.f / 512.f);
    float var = ssq[0] * (1.f / 512.f) - mu * mu;
    float rs  = rsqrtf(var + 1e-5f);
    {
        float nrm = (v0 - mu) * rs * g[tid] + bb[tid];
        float s = nrm / (1.f + __expf(-nrm));
        out[(size_t)row * 512 + tid] = s + x[(size_t)row * 512 + tid];
    }
    {
        int j = tid + 256;
        float nrm = (v1 - mu) * rs * g[j] + bb[j];
        float s = nrm / (1.f + __expf(-nrm));
        out[(size_t)row * 512 + j] = s + x[(size_t)row * 512 + j];
    }
}

// ---------------------------------------------------------------- host launcher
extern "C" void kernel_launch(void* const* d_in, const int* in_sizes, int n_in,
                              void* d_out, int out_size, void* d_ws, size_t ws_size,
                              hipStream_t stream) {
    (void)in_sizes; (void)n_in; (void)out_size; (void)ws_size;
    const float* x    = (const float*)d_in[0];
    const float* ln_g = (const float*)d_in[19];
    const float* ln_b = (const float*)d_in[20];

    char* ws = (char*)d_ws;
    auto alloc = [&](size_t bytes) { char* p = ws; ws += (bytes + 255) & ~(size_t)255; return p; };

    __bf16* xbf[2];
    xbf[0]        = (__bf16*)alloc((size_t)MROWS * D_MODEL * 2);
    xbf[1]        = (__bf16*)alloc((size_t)MROWS * D_MODEL * 2);
    __bf16* winbf = (__bf16*)alloc((size_t)2 * D_INNER * D_MODEL * 2);
    __bf16* wxbf  = (__bf16*)alloc((size_t)64 * D_INNER * 2);
    __bf16* wdtbf = (__bf16*)alloc((size_t)D_INNER * DT_RANK * 2);
    __bf16* wobf  = (__bf16*)alloc((size_t)D_MODEL * D_INNER * 2);
    float*  xz    = (float*) alloc((size_t)MROWS * 2 * D_INNER * 4);
    float*  uf    = (float*) alloc((size_t)MROWS * D_INNER * 4);
    __bf16* ubf   = (__bf16*)alloc((size_t)MROWS * D_INNER * 2);   // doubles as y_bf
    float*  proj  = (float*) alloc((size_t)MROWS * 64 * 4);
    __bf16* pdtbf = (__bf16*)alloc((size_t)MROWS * DT_RANK * 2);
    float*  odir[2];
    odir[0]       = (float*) alloc((size_t)MROWS * D_MODEL * 4);
    odir[1]       = (float*) alloc((size_t)MROWS * D_MODEL * 4);

    prep_x<<<(MROWS * D_MODEL) / 256, 256, 0, stream>>>(x, xbf[0], xbf[1]);

    for (int dir = 0; dir < 2; ++dir) {
        const float* W_in   = (const float*)d_in[1 + 9 * dir + 0];
        const float* conv_w = (const float*)d_in[1 + 9 * dir + 1];
        const float* conv_b = (const float*)d_in[1 + 9 * dir + 2];
        const float* W_xp   = (const float*)d_in[1 + 9 * dir + 3];
        const float* W_dt   = (const float*)d_in[1 + 9 * dir + 4];
        const float* b_dt   = (const float*)d_in[1 + 9 * dir + 5];
        const float* A_log  = (const float*)d_in[1 + 9 * dir + 6];
        const float* D_skip = (const float*)d_in[1 + 9 * dir + 7];
        const float* W_out  = (const float*)d_in[1 + 9 * dir + 8];

        cvt_f32_bf16<<<(2 * D_INNER * D_MODEL + 255) / 256, 256, 0, stream>>>(W_in, winbf, 2 * D_INNER * D_MODEL);
        cvt_f32_bf16<<<(64 * D_INNER + 255) / 256, 256, 0, stream>>>(W_xp, wxbf, 64 * D_INNER);
        cvt_f32_bf16<<<(D_INNER * DT_RANK + 255) / 256, 256, 0, stream>>>(W_dt, wdtbf, D_INNER * DT_RANK);
        cvt_f32_bf16<<<(D_MODEL * D_INNER + 255) / 256, 256, 0, stream>>>(W_out, wobf, D_MODEL * D_INNER);

        // xz = x @ W_in^T        [4096 x 2048], K=512
        gemm_bf16_nt<<<dim3((2 * D_INNER) / 64, MROWS / 128), 256, 0, stream>>>(
            xbf[dir], D_MODEL, winbf, D_MODEL, xz, 2 * D_INNER, D_MODEL);
        // u = silu(conv(xz[:, :1024]))
        conv_silu<<<(MROWS * D_INNER) / 256, 256, 0, stream>>>(xz, conv_w, conv_b, uf, ubf);
        // proj = u @ W_xproj^T   [4096 x 64], K=1024
        gemm_bf16_nt<<<dim3(1, MROWS / 128), 256, 0, stream>>>(
            ubf, D_INNER, wxbf, D_INNER, proj, 64, D_INNER);
        cvt_projdt<<<(MROWS * DT_RANK) / 256, 256, 0, stream>>>(proj, pdtbf);
        // dtraw = proj_dt @ W_dt^T [4096 x 1024], K=32 -> strided into xz[:, :1024]
        gemm_bf16_nt<<<dim3(D_INNER / 64, MROWS / 128), 256, 0, stream>>>(
            pdtbf, DT_RANK, wdtbf, DT_RANK, xz, 2 * D_INNER, DT_RANK);
        // selective scan (fused softplus, D-skip, z-gate), writes y_bf
        scan_kernel<<<dim3(D_INNER / 256, NBATCH), 256, 0, stream>>>(
            xz, proj, uf, b_dt, A_log, D_skip, ubf);
        // out_dir = y @ W_out^T  [4096 x 512], K=1024
        gemm_bf16_nt<<<dim3(D_MODEL / 64, MROWS / 128), 256, 0, stream>>>(
            ubf, D_INNER, wobf, D_INNER, odir[dir], D_MODEL, D_INNER);
    }

    combine_ln<<<MROWS, 256, 0, stream>>>(odir[0], odir[1], x, ln_g, ln_b, (float*)d_out);
}
